// CombinedSurvLoss_1803886265545
// MI455X (gfx1250) — compile-verified, matrix operand order checked by reading, not run on hardware
//
#include <hip/hip_runtime.h>

// Constants from the reference
#define ALPHA_   0.15f
#define RANK_W_  0.1f
#define EPS_     1e-7f
#define JCHUNK   512

typedef float    v2f  __attribute__((ext_vector_type(2)));
typedef float    v8f  __attribute__((ext_vector_type(8)));
typedef _Float16 v16h __attribute__((ext_vector_type(16)));

#if defined(__has_builtin)
#  if __has_builtin(__builtin_amdgcn_wmma_f32_16x16x4_f32)
#    define USE_F32_WMMA 1
#  endif
#endif
#ifndef USE_F32_WMMA
#  define USE_F32_WMMA 0
#endif

// ---------------------------------------------------------------------------
// ws layout (floats): [0]=nll_sum  [1]=rank_total  [2]=rank_count  [3..7]=pad
//                     [8 .. 8+B)  = risk[i]
// ---------------------------------------------------------------------------

__global__ void init_ws_kernel(float* ws) {
    if (threadIdx.x < 4) ws[threadIdx.x] = 0.0f;
}

// Per-sample NLL + risk computation. One thread per sample.
__global__ __launch_bounds__(256) void prep_kernel(
    const float* __restrict__ hz, const float* __restrict__ S,
    const int* __restrict__ Y, const int* __restrict__ c,
    float* __restrict__ ws, float* __restrict__ risk, int B) {
    int i = blockIdx.x * 256 + threadIdx.x;
    float loss = 0.0f;
    if (i < B) {
        float4 h4 = reinterpret_cast<const float4*>(hz)[i];
        risk[i] = h4.x + h4.y + h4.z + h4.w;

        int   yi = Y[i];
        float cf = (float)c[i];
        const float* hr = hz + (size_t)i * 4;
        const float* sr = S  + (size_t)i * 4;
        float s_prev = (yi == 0) ? 1.0f : sr[yi - 1];
        s_prev       = fmaxf(s_prev, EPS_);
        float hh     = fmaxf(hr[yi], EPS_);
        float s_now  = fmaxf(sr[yi], EPS_);
        float unc = -(1.0f - cf) * (logf(s_prev) + logf(hh));
        float cen = -cf * logf(s_now);
        loss = (1.0f - ALPHA_) * (unc + cen) + ALPHA_ * unc;
    }
    // wave32 reduce then cross-wave via LDS
    #pragma unroll
    for (int off = 16; off > 0; off >>= 1) loss += __shfl_down(loss, off, 32);
    __shared__ float red[8];
    int lane = threadIdx.x & 31, wid = threadIdx.x >> 5;
    if (lane == 0) red[wid] = loss;
    __syncthreads();
    if (threadIdx.x == 0) {
        float s = 0.0f;
        #pragma unroll
        for (int r = 0; r < 8; ++r) s += red[r];
        atomicAdd(&ws[0], s);
    }
}

// Pairwise ranking: each wave owns 16 i-rows; j swept via LDS in chunks.
// Sum over j is folded into WMMA with an all-ones B matrix:
//   acc_t accumulates sum_j mask*relu(risk_j - risk_i)  (16 i-rows at once)
//   acc_c accumulates sum_j mask                        (pair count)
// Requires B % 128 == 0 and B % JCHUNK == 0 (true for B=8192).
__global__ __launch_bounds__(256) void pair_kernel(
    const int* __restrict__ Y, const int* __restrict__ c,
    const float* __restrict__ risk, float* __restrict__ ws, int B) {
    __shared__ float s_risk[JCHUNK];
    __shared__ int   s_y[JCHUNK];

    const int lane = threadIdx.x & 31;
    const int wid  = threadIdx.x >> 5;
    const int m    = lane & 15;     // i-row within the wave's 16x tile
    const int ktop = lane >> 4;     // 0: K lanes 0..15 half, 1: upper half
    const int i    = blockIdx.x * 128 + wid * 16 + m;

    const float ri  = risk[i];
    const int   yi  = Y[i];
    const float ci0 = (c[i] == 0) ? 1.0f : 0.0f;

    v8f acc_t = {};
    v8f acc_c = {};

#if USE_F32_WMMA
    v2f bones; bones.x = 1.0f; bones.y = 1.0f;
#else
    v16h hones;
    #pragma unroll
    for (int e = 0; e < 16; ++e) hones[e] = (_Float16)1.0f;
#endif

    for (int j0 = 0; j0 < B; j0 += JCHUNK) {
        __syncthreads();
        // stage next j chunk; also prefetch the chunk after (clamped, no branch)
        int pbase = (j0 + JCHUNK < B) ? (j0 + JCHUNK) : 0;
        __builtin_prefetch(risk + pbase + (int)threadIdx.x, 0, 1);
        for (int t = threadIdx.x; t < JCHUNK; t += 256) {
            s_risk[t] = risk[j0 + t];
            s_y[t]    = Y[j0 + t];
        }
        __syncthreads();

#if USE_F32_WMMA
        // A is 16x4 f32: lane (ktop*16+m) holds K = ktop*2 + {0,1} in {a.x,a.y}
        for (int jj = 0; jj < JCHUNK; jj += 4) {
            int k = jj + ktop * 2;
            float r0 = s_risk[k],     r1 = s_risk[k + 1];
            int   y0 = s_y[k],        y1 = s_y[k + 1];
            v2f a, am;
            am.x = (y0 > yi) ? ci0 : 0.0f;
            am.y = (y1 > yi) ? ci0 : 0.0f;
            a.x  = am.x * fmaxf(r0 - ri, 0.0f);
            a.y  = am.y * fmaxf(r1 - ri, 0.0f);
            acc_t = __builtin_amdgcn_wmma_f32_16x16x4_f32(
                false, a, false, bones, (short)0, acc_t, false, false);
            acc_c = __builtin_amdgcn_wmma_f32_16x16x4_f32(
                false, am, false, bones, (short)0, acc_c, false, false);
        }
#else
        // Fallback: f16 A 16x32. lane(ktop,m): element e<8 -> K=ktop*8+e,
        // e>=8 -> K=16+ktop*8+(e-8).
        for (int jj = 0; jj < JCHUNK; jj += 32) {
            v16h a, am;
            #pragma unroll
            for (int e = 0; e < 16; ++e) {
                int kk = jj + ((e < 8) ? (ktop * 8 + e)
                                       : (16 + ktop * 8 + (e - 8)));
                float rj  = s_risk[kk];
                int   yj  = s_y[kk];
                float msk = (yj > yi) ? ci0 : 0.0f;
                am[e] = (_Float16)msk;
                a[e]  = (_Float16)(msk * fmaxf(rj - ri, 0.0f));
            }
            acc_t = __builtin_amdgcn_wmma_f32_16x16x32_f16(
                false, a, false, hones, (short)0, acc_t, false, false);
            acc_c = __builtin_amdgcn_wmma_f32_16x16x32_f16(
                false, am, false, hones, (short)0, acc_c, false, false);
        }
#endif
    }

    // D columns are identical (B = ones). Per-lane sum over 8 VGPR rows, then
    // full 32-lane xor reduce = 16 * (rows0-7 + rows8-15); scale by 1/16.
    float st = 0.0f, sc = 0.0f;
    #pragma unroll
    for (int r = 0; r < 8; ++r) { st += acc_t[r]; sc += acc_c[r]; }
    #pragma unroll
    for (int off = 16; off > 0; off >>= 1) {
        st += __shfl_xor(st, off, 32);
        sc += __shfl_xor(sc, off, 32);
    }
    if (lane == 0) {
        atomicAdd(&ws[1], st * (1.0f / 16.0f));
        atomicAdd(&ws[2], sc * (1.0f / 16.0f));
    }
}

__global__ void combine_kernel(const float* __restrict__ ws,
                               float* __restrict__ out, int B) {
    if (threadIdx.x == 0 && blockIdx.x == 0) {
        float nll   = ws[0] / (float)B;
        float total = ws[1];
        float count = ws[2];
        float rank  = (count > 0.0f) ? total / fmaxf(count, 1.0f) : total;
        out[0] = nll + RANK_W_ * rank;
    }
}

extern "C" void kernel_launch(void* const* d_in, const int* in_sizes, int n_in,
                              void* d_out, int out_size, void* d_ws, size_t ws_size,
                              hipStream_t stream) {
    const float* hz = (const float*)d_in[0];
    const float* S  = (const float*)d_in[1];
    const int*   Y  = (const int*)d_in[2];
    const int*   c  = (const int*)d_in[3];
    float* ws   = (float*)d_ws;
    float* risk = ws + 8;
    float* out  = (float*)d_out;
    int B = in_sizes[2];   // 8192

    init_ws_kernel<<<1, 32, 0, stream>>>(ws);
    prep_kernel<<<(B + 255) / 256, 256, 0, stream>>>(hz, S, Y, c, ws, risk, B);
    pair_kernel<<<B / 128, 256, 0, stream>>>(Y, c, risk, ws, B);
    combine_kernel<<<1, 32, 0, stream>>>(ws, out, B);
}
